// RoIHeads_15607911153786
// MI455X (gfx1250) — compile-verified, matrix operand order checked by reading, not run on hardware
//
#include <hip/hip_runtime.h>
#include <hip/hip_bf16.h>

// Problem constants (match reference)
#define BIMG   16
#define NPROP  4096
#define NCLS   61
#define KDET   100
#define MAXV   24576           // candidate cap per image (LDS-resident scores)
#define IMGW   1333.0f
#define IMGH   800.0f
#define DWCLIP 4.135166556742356f   // log(1000/16)

typedef __attribute__((ext_vector_type(16))) _Float16 v16h;
typedef __attribute__((ext_vector_type(8)))  float    v8f;

__device__ __forceinline__ float clampf(float x, float lo, float hi) {
    return fminf(fmaxf(x, lo), hi);
}

__global__ void init_kernel(int* cnt) {
    if (threadIdx.x < BIMG) cnt[threadIdx.x] = 0;
}

// Phase 1: softmax (WMMA row-sum) + box decode + filter + compaction.
// One wave handles a 16-row tile; lane L holds row (L&15), K-chunk (L>>4)
// following the CDNA5 f16 A-matrix 16x32 layout.
__global__ __launch_bounds__(256) void score_decode_kernel(
    const float* __restrict__ logits, const float* __restrict__ reg,
    const float* __restrict__ props,  int* __restrict__ cnt,
    float4* __restrict__ cbox, float4* __restrict__ cmeta, int maxv)
{
    const int lane = threadIdx.x & 31;
    const int wave = threadIdx.x >> 5;
    const int tile = blockIdx.x * 8 + wave;
    const int m    = lane & 15;
    const int half = lane >> 4;
    const int r    = tile * 16 + m;          // global row (B*N rows total)
    const int b    = r >> 12;                // image index (N = 4096)
    const int n    = r & (NPROP - 1);

    const float* lrow = logits + (long)r * NCLS;

    // Load this lane's 32 logits per the A-matrix layout; track max.
    float v0[16], v1[16];
    float mx = -INFINITY;
#pragma unroll
    for (int j = 0; j < 16; ++j) {
        int k = (j < 8) ? (half * 8 + j) : (16 + half * 8 + (j - 8));
        float x = lrow[k];                   // k <= 31 < 61 always
        v0[j] = x; mx = fmaxf(mx, x);
    }
#pragma unroll
    for (int j = 0; j < 16; ++j) {
        int k = 32 + ((j < 8) ? (half * 8 + j) : (16 + half * 8 + (j - 8)));
        float x = (k < NCLS) ? lrow[k] : -INFINITY;   // pad classes -> exp 0
        v1[j] = x; mx = fmaxf(mx, x);
    }
    // Row m is split across lanes m and m+16 -> combine for row max.
    mx = fmaxf(mx, __shfl_xor(mx, 16));

    // exp (stable); pack f16 A tiles for the WMMA row-sum.
    float e0[16], e1[16];
    v16h a0, a1, bones;
#pragma unroll
    for (int j = 0; j < 16; ++j) {
        float e = __expf(0.0f); // placeholder removed below
        (void)e;
        float ea = expf(v0[j] - mx); e0[j] = ea; a0[j] = (_Float16)ea;
        float eb = expf(v1[j] - mx); e1[j] = eb; a1[j] = (_Float16)eb;
        bones[j] = (_Float16)1.0f;           // B = ones(32x16), layout-invariant
    }

    // D[m,n] = sum_k A[m,k] : softmax denominator on the matrix pipe.
    v8f c = {};
    c = __builtin_amdgcn_wmma_f32_16x16x32_f16(false, a0, false, bones,
                                               (short)0, c, false, false);
    c = __builtin_amdgcn_wmma_f32_16x16x32_f16(false, a1, false, bones,
                                               (short)0, c, false, false);

    // Broadcast row sums: lane 0 holds rows 0..7, lane 16 rows 8..15 (N=0 col).
    __shared__ float rs[8][16];
    if ((lane & 15) == 0) {
#pragma unroll
        for (int q = 0; q < 8; ++q) rs[wave][half * 8 + q] = c[q];
    }
    __syncthreads();
    const float inv = 1.0f / rs[wave][m];

    // Decode/filter this lane's (row, class) slots; compact valid candidates.
    const float4 p = reinterpret_cast<const float4*>(props)[r];  // [B*N][4]
    const float w  = p.z - p.x, h = p.w - p.y;
    const float cx = p.x + 0.5f * w, cy = p.y + 0.5f * h;
    const float4* rrow = reinterpret_cast<const float4*>(reg) + (long)r * NCLS;

    auto emit = [&](int k, float e) {
        if (k <= 0 || k >= NCLS) return;     // drop background + pad
        float s  = e * inv;
        float4 t = rrow[k];
        float dx = t.x / 10.0f, dy = t.y / 10.0f;
        float dw = fminf(t.z / 5.0f, DWCLIP);
        float dh = fminf(t.w / 5.0f, DWCLIP);
        float pcx = dx * w + cx, pcy = dy * h + cy;
        float pw = expf(dw) * w, ph = expf(dh) * h;
        float x1 = clampf(pcx - 0.5f * pw, 0.f, IMGW);
        float y1 = clampf(pcy - 0.5f * ph, 0.f, IMGH);
        float x2 = clampf(pcx + 0.5f * pw, 0.f, IMGW);
        float y2 = clampf(pcy + 0.5f * ph, 0.f, IMGH);
        if (s > 0.05f && (x2 - x1) >= 0.01f && (y2 - y1) >= 0.01f) {
            int pos = atomicAdd(&cnt[b], 1);
            if (pos < maxv) {
                size_t o = (size_t)b * maxv + pos;
                cbox[o] = make_float4(x1, y1, x2, y2);
                int flat = n * (NCLS - 1) + (k - 1);     // reference flat index
                // key = flat*64 + label  (< 2^24, exact in f32): deterministic
                // tie-break + label recovery, independent of compaction order.
                cmeta[o] = make_float4(s, (float)k, (float)(flat * 64 + k), 0.f);
            }
        }
    };
#pragma unroll
    for (int j = 0; j < 16; ++j) {
        int k = (j < 8) ? (half * 8 + j) : (16 + half * 8 + (j - 8));
        emit(k,      e0[j]);
        emit(k + 32, e1[j]);
    }
}

// Phase 2: greedy class-aware NMS, one workgroup per image.
// Mutable scores + keys live in 192 KB of LDS (CDNA5: 320 KB/WGP).
// Each round fuses suppress-by-previous-winner with the argmax sweep.
__global__ __launch_bounds__(1024) void nms_kernel(
    const int* __restrict__ cnt, const float4* __restrict__ cbox,
    const float4* __restrict__ cmeta, float* __restrict__ out, int maxv)
{
    extern __shared__ char smem[];
    float* s_sc  = (float*)smem;          // MAXV scores (mutable)
    float* s_key = s_sc + MAXV;           // MAXV tie-break keys (flat*64+label)
    __shared__ float  red_s[1024];
    __shared__ float  red_k[1024];
    __shared__ int    red_i[1024];
    __shared__ float4 selb;
    __shared__ int    plab;
    __shared__ int    alive;

    const int b = blockIdx.x, tid = threadIdx.x;
    int V = cnt[b]; if (V > maxv) V = maxv;
    const float4* boxes = cbox + (size_t)b * maxv;
    const float4* metas = cmeta + (size_t)b * maxv;

    for (int i = tid; i < V; i += 1024) {
        float4 mt = metas[i];
        s_sc[i]  = mt.x;
        s_key[i] = mt.z;
    }
    if (tid == 0) alive = 0;
    __syncthreads();

    float* ob  = out;                       // [B][K][4]
    float* osc = out + BIMG * KDET * 4;     // [B][K]
    int*   olb = (int*)(out + BIMG * KDET * 5); // [B][K] int32 bits

    for (int it = 0; it < KDET; ++it) {
        const int havep = alive;
        float4 pb; float pa1 = 0.f; int pl = 0;
        if (havep) { pb = selb; pl = plab; pa1 = (pb.z - pb.x) * (pb.w - pb.y); }

        float bs = -2.f, bk = 3.0e38f; int bi = -1;
        for (int i = tid; i < V; i += 1024) {
            if (i + 1024 < V) __builtin_prefetch(boxes + i + 1024, 0, 1);
            float s   = s_sc[i];
            float key = s_key[i];
            if (havep && s > 0.f && ((((int)key) & 63) == pl)) {
                // same label -> real IoU test (offset trick is equivalent)
                float4 bx = boxes[i];
                float ix = fminf(pb.z, bx.z) - fmaxf(pb.x, bx.x);
                float iy = fminf(pb.w, bx.w) - fmaxf(pb.y, bx.y);
                float inter = fmaxf(ix, 0.f) * fmaxf(iy, 0.f);
                float a2 = (bx.z - bx.x) * (bx.w - bx.y);
                float iou = inter / (pa1 + a2 - inter);
                if (iou > 0.5f) { s = -1.f; s_sc[i] = s; }  // self included
            }
            if (s > bs || (s == bs && key < bk)) { bs = s; bk = key; bi = i; }
        }
        red_s[tid] = bs; red_k[tid] = bk; red_i[tid] = bi;
        __syncthreads();
        for (int off = 512; off > 0; off >>= 1) {
            if (tid < off) {
                float s2 = red_s[tid + off];
                if (s2 > red_s[tid] ||
                    (s2 == red_s[tid] && red_k[tid + off] < red_k[tid])) {
                    red_s[tid] = s2;
                    red_k[tid] = red_k[tid + off];
                    red_i[tid] = red_i[tid + off];
                }
            }
            __syncthreads();
        }
        if (tid == 0) {
            int win = red_i[0]; float sw = red_s[0];
            int o = b * KDET + it;
            if (win >= 0 && sw > 0.f) {
                float4 bx = boxes[win];
                int lab = ((int)red_k[0]) & 63;
                ob[o * 4 + 0] = bx.x; ob[o * 4 + 1] = bx.y;
                ob[o * 4 + 2] = bx.z; ob[o * 4 + 3] = bx.w;
                osc[o] = sw; olb[o] = lab;
                selb = bx; plab = lab; alive = 1;
            } else {
                ob[o * 4 + 0] = 0.f; ob[o * 4 + 1] = 0.f;
                ob[o * 4 + 2] = 0.f; ob[o * 4 + 3] = 0.f;
                osc[o] = 0.f; olb[o] = 0; alive = 0;
            }
        }
        __syncthreads();
    }
}

extern "C" void kernel_launch(void* const* d_in, const int* in_sizes, int n_in,
                              void* d_out, int out_size, void* d_ws, size_t ws_size,
                              hipStream_t stream) {
    const float* logits = (const float*)d_in[0];   // [B*N, C]
    const float* reg    = (const float*)d_in[1];   // [B*N, C*4]
    const float* props  = (const float*)d_in[2];   // [B, N, 4]
    float* out = (float*)d_out;                    // boxes|scores|labels

    int maxv = MAXV;
    size_t need = 256 + (size_t)BIMG * (size_t)maxv * 32;
    if (ws_size < need) {
        size_t avail = (ws_size > 256) ? (ws_size - 256) / ((size_t)BIMG * 32) : 0;
        maxv = (int)((avail < (size_t)MAXV) ? avail : (size_t)MAXV);
    }

    int*    cnt   = (int*)d_ws;
    float4* cbox  = (float4*)((char*)d_ws + 256);
    float4* cmeta = cbox + (size_t)BIMG * (size_t)maxv;

    init_kernel<<<1, 32, 0, stream>>>(cnt);
    score_decode_kernel<<<(BIMG * NPROP) / 128, 256, 0, stream>>>(
        logits, reg, props, cnt, cbox, cmeta, maxv);
    nms_kernel<<<BIMG, 1024, (size_t)MAXV * 2 * sizeof(float), stream>>>(
        cnt, cbox, cmeta, out, maxv);
}
